// GATLayer_72636486910245
// MI455X (gfx1250) — matrix operand order repaired and checked by reading, not compiled, not run on hardware
//
#include <hip/hip_runtime.h>
#include <hip/hip_bf16.h>

typedef __attribute__((ext_vector_type(16))) _Float16 v16h;
typedef __attribute__((ext_vector_type(8)))  _Float16 v8h;
typedef __attribute__((ext_vector_type(8)))  float    v8f;

#define N_NODES 4096
#define FEAT    256
#define HEADS   8
#define HDIM    32
#define LN_EPS  1e-5f

// ---------------- workspace layout (bytes) ----------------
#define WS_XH   ((size_t)0)                         // 4096*256 f16 = 2 MB
#define WS_QH   (WS_XH  + (size_t)N_NODES*FEAT*2)   // q (pre-scaled) f16
#define WS_KH   (WS_QH  + (size_t)N_NODES*FEAT*2)   // k f16
#define WS_VT   (WS_KH  + (size_t)N_NODES*FEAT*2)   // v transposed: [h][d][n] f16
#define WS_MSG  (WS_VT  + (size_t)N_NODES*FEAT*2)   // attention output f16
#define WS_WT   (WS_MSG + (size_t)N_NODES*FEAT*2)   // [Wq|Wk|Wv]^T f16: [768][256]
#define WS_WOT  (WS_WT  + (size_t)768*FEAT*2)       // Wo^T f16: [256][256]

// ---------------- WMMA operand loaders ----------------
// A-matrix 16x32 f16: lane<16 -> M=lane, K=0..15;
//                     lane>=16 -> M=lane-16, K=16..31
__device__ __forceinline__ v16h load_A(const _Float16* base, int row0, int ld,
                                       int k0, int lane) {
    int m = lane & 15, hl = lane >> 4;
    const _Float16* p = base + (size_t)(row0 + m) * ld + k0;
    v8h c0 = *(const v8h*)(p + 16 * hl);
    v8h c1 = *(const v8h*)(p + 16 * hl + 8);
    v16h a;
#pragma unroll
    for (int i = 0; i < 8; ++i) { a[i] = c0[i]; a[8 + i] = c1[i]; }
    return a;
}

// B-matrix 32x16 f16 supplied from N-major storage Bt[n][k] (k contiguous):
// lane<16 -> N=lane, K=0..15; lane>=16 -> N=lane-16, K=16..31
__device__ __forceinline__ v16h load_Bt(const _Float16* base, int col0, int ld,
                                        int k0, int lane) {
    int n = lane & 15, hl = lane >> 4;
    const _Float16* p = base + (size_t)(col0 + n) * ld + k0 + 16 * hl;
    v8h c0 = *(const v8h*)(p);
    v8h c1 = *(const v8h*)(p + 8);
    v16h b;
#pragma unroll
    for (int i = 0; i < 8; ++i) { b[i] = c0[i]; b[8 + i] = c1[i]; }
    return b;
}

__device__ __forceinline__ unsigned pack2h(float a, float b) {
    union { _Float16 h[2]; unsigned u; } t;
    t.h[0] = (_Float16)a; t.h[1] = (_Float16)b;
    return t.u;
}

// ---------------- prep kernels ----------------
__global__ __launch_bounds__(256) void cvt_x_kernel(const float* __restrict__ x,
                                                    _Float16* __restrict__ xh) {
    int i = blockIdx.x * 256 + threadIdx.x;
    xh[i] = (_Float16)x[i];
}

__global__ __launch_bounds__(256) void prep_w_kernel(
    const float* __restrict__ Wq, const float* __restrict__ Wk,
    const float* __restrict__ Wv, const float* __restrict__ Wo,
    _Float16* __restrict__ Wt, _Float16* __restrict__ Wot) {
    int idx = blockIdx.x * 256 + threadIdx.x;   // 0 .. 1024*256-1
    int j = idx >> 8, i = idx & 255;            // j = output col, i = input k
    if (j < 256)       Wt[(size_t)j * 256 + i]          = (_Float16)Wq[(size_t)i * 256 + j];
    else if (j < 512)  Wt[(size_t)j * 256 + i]          = (_Float16)Wk[(size_t)i * 256 + (j - 256)];
    else if (j < 768)  Wt[(size_t)j * 256 + i]          = (_Float16)Wv[(size_t)i * 256 + (j - 512)];
    else               Wot[(size_t)(j - 768) * 256 + i] = (_Float16)Wo[(size_t)i * 256 + (j - 768)];
}

// ---------------- fused QKV projection (2x2 register-blocked) ----------------
// One wave per 32x32 output tile of [q|k|v] (cols 0..767).
// q stored pre-scaled (folds 1/sqrt(D)). v stored transposed [h][d][n].
__device__ __forceinline__ void store_qkv_tile(
    const v8f& acc, int row0, int col0, int lane,
    const float* __restrict__ bq, const float* __restrict__ bk,
    const float* __restrict__ bv, float scale,
    _Float16* __restrict__ qh, _Float16* __restrict__ kh,
    _Float16* __restrict__ vT) {
    int n = lane & 15, hl = lane >> 4;
    int col = col0 + n;                     // 16-wide tile never straddles a 256 segment
    if (col < 512) {                        // q or k: row-major store
        float bias = (col < 256) ? bq[col] : bk[col - 256];
        float sc = (col < 256) ? scale : 1.0f;
        _Float16* dst = (col < 256) ? qh : kh;
        int c = col & 255;
#pragma unroll
        for (int r = 0; r < 8; ++r) {
            int row = row0 + r + 8 * hl;
            dst[(size_t)row * FEAT + c] = (_Float16)((acc[r] + bias) * sc);
        }
    } else {                                // v: transposed store vT[h][d][n]
        int cv = col - 512;
        float bias = bv[cv];
        _Float16* dst = vT + (size_t)cv * N_NODES + row0 + 8 * hl;
        v8h out;
#pragma unroll
        for (int r = 0; r < 8; ++r) out[r] = (_Float16)(acc[r] + bias);
        *(v8h*)dst = out;                   // 8 consecutive nodes, 16B store
    }
}

__global__ __launch_bounds__(32) void qkv_gemm_kernel(
    const _Float16* __restrict__ xh, const _Float16* __restrict__ Wt,
    const float* __restrict__ bq, const float* __restrict__ bk,
    const float* __restrict__ bv, const float* __restrict__ scale_p,
    _Float16* __restrict__ qh, _Float16* __restrict__ kh,
    _Float16* __restrict__ vT) {
    int lane = threadIdx.x;
    int col0 = blockIdx.x * 32;   // 0..767
    int row0 = blockIdx.y * 32;   // 0..4095
    v8f a00 = {}, a01 = {}, a10 = {}, a11 = {};
#pragma unroll
    for (int k0 = 0; k0 < FEAT; k0 += 32) {
        v16h a0 = load_A(xh, row0,      FEAT, k0, lane);
        v16h a1 = load_A(xh, row0 + 16, FEAT, k0, lane);
        v16h b0 = load_Bt(Wt, col0,      FEAT, k0, lane);
        v16h b1 = load_Bt(Wt, col0 + 16, FEAT, k0, lane);
        a00 = __builtin_amdgcn_wmma_f32_16x16x32_f16(false, a0, false, b0, (short)0, a00, false, false);
        a01 = __builtin_amdgcn_wmma_f32_16x16x32_f16(false, a0, false, b1, (short)0, a01, false, false);
        a10 = __builtin_amdgcn_wmma_f32_16x16x32_f16(false, a1, false, b0, (short)0, a10, false, false);
        a11 = __builtin_amdgcn_wmma_f32_16x16x32_f16(false, a1, false, b1, (short)0, a11, false, false);
    }
    float scale = *scale_p;
    store_qkv_tile(a00, row0,      col0,      lane, bq, bk, bv, scale, qh, kh, vT);
    store_qkv_tile(a01, row0,      col0 + 16, lane, bq, bk, bv, scale, qh, kh, vT);
    store_qkv_tile(a10, row0 + 16, col0,      lane, bq, bk, bv, scale, qh, kh, vT);
    store_qkv_tile(a11, row0 + 16, col0 + 16, lane, bq, bk, bv, scale, qh, kh, vT);
}

// ---------------- flash attention (32 queries / wave) ----------------
// S^T = K*Q^T puts each query's scores in a lane pair (lane n: keys r+8*0,
// lane n+16: keys r+8*1), so softmax needs only in-lane ops + 1 shfl_xor(16).
// P^T is repacked into the B-operand layout with one shfl_xor(16) exchange,
// then O^T = V^T * P^T.
__device__ __forceinline__ v16h softmax_pack(const v8f& s_lo, const v8f& s_hi,
                                             float& m_run, float& l_run,
                                             float& corr, int hl) {
    float tmax = fmaxf(s_lo[0], s_hi[0]);
#pragma unroll
    for (int r = 1; r < 8; ++r) tmax = fmaxf(tmax, fmaxf(s_lo[r], s_hi[r]));
    tmax = fmaxf(tmax, __shfl_xor(tmax, 16, 32));
    float m_new = fmaxf(m_run, tmax);
    corr = __expf(m_run - m_new);
    float p0[8], p1[8], psum = 0.0f;
#pragma unroll
    for (int r = 0; r < 8; ++r) {
        p0[r] = __expf(s_lo[r] - m_new);
        p1[r] = __expf(s_hi[r] - m_new);
        psum += p0[r] + p1[r];
    }
    psum += __shfl_xor(psum, 16, 32);
    l_run = l_run * corr + psum;
    m_run = m_new;
    unsigned pk0[4], pk1[4], op0[4], op1[4];
#pragma unroll
    for (int i = 0; i < 4; ++i) {
        pk0[i] = pack2h(p0[2 * i], p0[2 * i + 1]);
        pk1[i] = pack2h(p1[2 * i], p1[2 * i + 1]);
    }
#pragma unroll
    for (int i = 0; i < 4; ++i) {
        op0[i] = __shfl_xor(pk0[i], 16, 32);
        op1[i] = __shfl_xor(pk1[i], 16, 32);
    }
    union { v16h v; unsigned u[8]; } bb;
#pragma unroll
    for (int i = 0; i < 4; ++i) {
        bb.u[i]     = hl ? op1[i] : pk0[i];   // keys 0-7  / 16-23
        bb.u[4 + i] = hl ? pk1[i] : op0[i];   // keys 8-15 / 24-31
    }
    return bb.v;
}

__global__ __launch_bounds__(32) void attn_kernel(
    const _Float16* __restrict__ qh, const _Float16* __restrict__ kh,
    const _Float16* __restrict__ vT, _Float16* __restrict__ msg) {
    int lane = threadIdx.x;
    int h  = blockIdx.y;
    int i0 = blockIdx.x * 32;               // query tile base (32 queries)
    int n = lane & 15, hl = lane >> 4;
    int m = lane & 15;

    // B operands: Q^T (dims x queries), loaded once (q already scaled)
    v16h qb0 = load_Bt(qh, i0,      FEAT, h * HDIM, lane);
    v16h qb1 = load_Bt(qh, i0 + 16, FEAT, h * HDIM, lane);

    const _Float16* vbase = vT + (size_t)h * HDIM * N_NODES;

    float m0 = -3.0e38f, l0 = 0.0f, m1 = -3.0e38f, l1 = 0.0f;
    v8f o00 = {}, o01 = {}, o10 = {}, o11 = {};  // [qtile][dims 0-15 / 16-31]
    v8f zero = {};

    for (int j0 = 0; j0 < N_NODES; j0 += 32) {
        // prefetch next K/V block into near caches (global_prefetch_b8)
        int jn = (j0 + 32 < N_NODES) ? j0 + 32 : j0;
        __builtin_prefetch(kh + (size_t)(jn + m) * FEAT + h * HDIM, 0, 3);
        __builtin_prefetch(vbase + (size_t)m * N_NODES + jn, 0, 3);

        // --- scores: S^T tiles, keys [j0,j0+16) / [j0+16,j0+32) ---
        v16h ka0 = load_A(kh, j0,      FEAT, h * HDIM, lane);
        v16h ka1 = load_A(kh, j0 + 16, FEAT, h * HDIM, lane);
        v8f s00 = __builtin_amdgcn_wmma_f32_16x16x32_f16(false, ka0, false, qb0, (short)0, zero, false, false);
        v8f s10 = __builtin_amdgcn_wmma_f32_16x16x32_f16(false, ka1, false, qb0, (short)0, zero, false, false);
        v8f s01 = __builtin_amdgcn_wmma_f32_16x16x32_f16(false, ka0, false, qb1, (short)0, zero, false, false);
        v8f s11 = __builtin_amdgcn_wmma_f32_16x16x32_f16(false, ka1, false, qb1, (short)0, zero, false, false);

        // --- online softmax per query tile ---
        float c0, c1;
        v16h bb0 = softmax_pack(s00, s10, m0, l0, c0, hl);
        v16h bb1 = softmax_pack(s01, s11, m1, l1, c1, hl);
#pragma unroll
        for (int r = 0; r < 8; ++r) {
            o00[r] *= c0; o01[r] *= c0;
            o10[r] *= c1; o11[r] *= c1;
        }

        // --- O^T += V^T * P^T ---
        v16h va0 = load_A(vbase, 0,  N_NODES, j0, lane);   // dims 0-15
        v16h va1 = load_A(vbase, 16, N_NODES, j0, lane);   // dims 16-31
        o00 = __builtin_amdgcn_wmma_f32_16x16x32_f16(false, va0, false, bb0, (short)0, o00, false, false);
        o01 = __builtin_amdgcn_wmma_f32_16x16x32_f16(false, va1, false, bb0, (short)0, o01, false, false);
        o10 = __builtin_amdgcn_wmma_f32_16x16x32_f16(false, va0, false, bb1, (short)0, o10, false, false);
        o11 = __builtin_amdgcn_wmma_f32_16x16x32_f16(false, va1, false, bb1, (short)0, o11, false, false);
    }

    float inv0 = 1.0f / l0;
    float inv1 = 1.0f / l1;
    _Float16* out0 = msg + (size_t)(i0 + n) * FEAT + h * HDIM;
    _Float16* out1 = msg + (size_t)(i0 + 16 + n) * FEAT + h * HDIM;
#pragma unroll
    for (int r = 0; r < 8; ++r) {
        out0[r + 8 * hl]      = (_Float16)(o00[r] * inv0);
        out0[16 + r + 8 * hl] = (_Float16)(o01[r] * inv0);
        out1[r + 8 * hl]      = (_Float16)(o10[r] * inv1);
        out1[16 + r + 8 * hl] = (_Float16)(o11[r] * inv1);
    }
}

// ------------- output projection + residual (2x2 register-blocked) -------------
__device__ __forceinline__ void store_y_tile(const v8f& acc, int row0, int col0,
                                             int lane, const float* __restrict__ bo,
                                             const float* __restrict__ x,
                                             float* __restrict__ y) {
    int n = lane & 15, hl = lane >> 4;
    int col = col0 + n;
    float bias = bo[col];
#pragma unroll
    for (int r = 0; r < 8; ++r) {
        int row = row0 + r + 8 * hl;
        y[(size_t)row * FEAT + col] = x[(size_t)row * FEAT + col] + acc[r] + bias;
    }
}

__global__ __launch_bounds__(32) void out_proj_kernel(
    const _Float16* __restrict__ msg, const _Float16* __restrict__ Wot,
    const float* __restrict__ bo, const float* __restrict__ x,
    float* __restrict__ y) {
    int lane = threadIdx.x;
    int col0 = blockIdx.x * 32;
    int row0 = blockIdx.y * 32;
    v8f a00 = {}, a01 = {}, a10 = {}, a11 = {};
#pragma unroll
    for (int k0 = 0; k0 < FEAT; k0 += 32) {
        v16h a0 = load_A(msg, row0,      FEAT, k0, lane);
        v16h a1 = load_A(msg, row0 + 16, FEAT, k0, lane);
        v16h b0 = load_Bt(Wot, col0,      FEAT, k0, lane);
        v16h b1 = load_Bt(Wot, col0 + 16, FEAT, k0, lane);
        a00 = __builtin_amdgcn_wmma_f32_16x16x32_f16(false, a0, false, b0, (short)0, a00, false, false);
        a01 = __builtin_amdgcn_wmma_f32_16x16x32_f16(false, a0, false, b1, (short)0, a01, false, false);
        a10 = __builtin_amdgcn_wmma_f32_16x16x32_f16(false, a1, false, b0, (short)0, a10, false, false);
        a11 = __builtin_amdgcn_wmma_f32_16x16x32_f16(false, a1, false, b1, (short)0, a11, false, false);
    }
    store_y_tile(a00, row0,      col0,      lane, bo, x, y);
    store_y_tile(a01, row0,      col0 + 16, lane, bo, x, y);
    store_y_tile(a10, row0 + 16, col0,      lane, bo, x, y);
    store_y_tile(a11, row0 + 16, col0 + 16, lane, bo, x, y);
}

// ---------------- in-place LayerNorm (one wave per row) ----------------
__global__ __launch_bounds__(32) void ln_kernel(float* __restrict__ y,
                                                const float* __restrict__ gamma,
                                                const float* __restrict__ beta) {
    int row = blockIdx.x;
    int lane = threadIdx.x;
    float* p = y + (size_t)row * FEAT + lane * 8;
    float v[8];
    float4 a = *(const float4*)(p);
    float4 b = *(const float4*)(p + 4);
    v[0]=a.x; v[1]=a.y; v[2]=a.z; v[3]=a.w;
    v[4]=b.x; v[5]=b.y; v[6]=b.z; v[7]=b.w;
    float s = 0.f;
#pragma unroll
    for (int i = 0; i < 8; ++i) s += v[i];
#pragma unroll
    for (int off = 16; off > 0; off >>= 1) s += __shfl_xor(s, off, 32);
    float mu = s * (1.0f / FEAT);
    float vs = 0.f;
#pragma unroll
    for (int i = 0; i < 8; ++i) { float d = v[i] - mu; vs += d * d; }
#pragma unroll
    for (int off = 16; off > 0; off >>= 1) vs += __shfl_xor(vs, off, 32);
    float rs = rsqrtf(vs * (1.0f / FEAT) + LN_EPS);
    int c0 = lane * 8;
#pragma unroll
    for (int i = 0; i < 8; ++i)
        p[i] = (v[i] - mu) * rs * gamma[c0 + i] + beta[c0 + i];
}

// ---------------- launcher ----------------
extern "C" void kernel_launch(void* const* d_in, const int* in_sizes, int n_in,
                              void* d_out, int out_size, void* d_ws, size_t ws_size,
                              hipStream_t stream) {
    const float* x     = (const float*)d_in[0];
    const float* Wq    = (const float*)d_in[1];
    const float* bq    = (const float*)d_in[2];
    const float* Wk    = (const float*)d_in[3];
    const float* bk    = (const float*)d_in[4];
    const float* Wv    = (const float*)d_in[5];
    const float* bv    = (const float*)d_in[6];
    const float* scale = (const float*)d_in[7];
    const float* Wo    = (const float*)d_in[8];
    const float* bo    = (const float*)d_in[9];
    const float* gamma = (const float*)d_in[10];
    const float* beta  = (const float*)d_in[11];

    char* ws = (char*)d_ws;
    _Float16* xh  = (_Float16*)(ws + WS_XH);
    _Float16* qh  = (_Float16*)(ws + WS_QH);
    _Float16* kh  = (_Float16*)(ws + WS_KH);
    _Float16* vT  = (_Float16*)(ws + WS_VT);
    _Float16* msg = (_Float16*)(ws + WS_MSG);
    _Float16* Wt  = (_Float16*)(ws + WS_WT);
    _Float16* Wot = (_Float16*)(ws + WS_WOT);
    float* y = (float*)d_out;

    cvt_x_kernel<<<N_NODES * FEAT / 256, 256, 0, stream>>>(x, xh);
    prep_w_kernel<<<1024, 256, 0, stream>>>(Wq, Wk, Wv, Wo, Wt, Wot);
    qkv_gemm_kernel<<<dim3(768 / 32, N_NODES / 32), 32, 0, stream>>>(
        xh, Wt, bq, bk, bv, scale, qh, kh, vT);
    attn_kernel<<<dim3(N_NODES / 32, HEADS), 32, 0, stream>>>(qh, kh, vT, msg);
    out_proj_kernel<<<dim3(FEAT / 32, N_NODES / 32), 32, 0, stream>>>(
        msg, Wot, bo, x, y);
    ln_kernel<<<N_NODES, 32, 0, stream>>>(y, gamma, beta);
}
